// GPT_73615739453489
// MI455X (gfx1250) — matrix operand order repaired and checked by reading, not compiled, not run on hardware
//
#include <hip/hip_runtime.h>
#include <hip/hip_bf16.h>
#include <math.h>

// ---------------------------------------------------------------------------
// GPT-2-ish forward (6 layers, D=1024, H=16, T=1024, B=2, V=50257) on gfx1250.
// All matmuls via v_wmma_f32_16x16x32_bf16 (fp32 -> bf16 RNE, f32 accumulate).
// Compute-bound (~570 GFLOP vs ~21us HBM traffic at 23.3 TB/s), so the GEMM
// mainloop is double-buffered through LDS: global prefetch of tile k+1
// overlaps the 8 WMMAs on tile k; one workgroup barrier per K-step.
// ---------------------------------------------------------------------------

#define NLAYER 6
#define NHEAD  16
#define DM     1024
#define HDIM   64
#define VOC    50257
#define NB     2
#define TS     1024
#define MR     (NB * TS)   // 2048 rows
#define LDSW   36          // LDS row stride in ushorts (8B-aligned frags, bank spread)

typedef __attribute__((ext_vector_type(16))) __bf16          v16bf;
typedef __attribute__((ext_vector_type(8)))  float           v8f;
typedef __attribute__((ext_vector_type(4)))  unsigned short  v4u;

union Frag {
  v4u            q[4];
  unsigned short s[16];
  v16bf          b;
};

__device__ __forceinline__ unsigned short f2bf(float x) {
  unsigned u = __builtin_bit_cast(unsigned, x);
  u += 0x7FFFu + ((u >> 16) & 1u);          // round-to-nearest-even
  return (unsigned short)(u >> 16);
}

__device__ __forceinline__ unsigned pack2bf(float x, float y) {
  unsigned a = __builtin_bit_cast(unsigned, x);
  unsigned b = __builtin_bit_cast(unsigned, y);
  a += 0x7FFFu + ((a >> 16) & 1u);
  b += 0x7FFFu + ((b >> 16) & 1u);
  return (a >> 16) | (b & 0xFFFF0000u);
}

__device__ __forceinline__ v8f wmma_bf16(const Frag& a, const Frag& b, v8f c) {
  return __builtin_amdgcn_wmma_f32_16x16x32_bf16(false, a.b, false, b.b,
                                                 (short)0, c, false, false);
}

// 16 consecutive floats (one half-row of a 128x32 staging tile) in registers.
struct Tile16 { float4 f[4]; };

__device__ __forceinline__ void ldg_tile(Tile16& t, const float* __restrict__ src) {
  #pragma unroll
  for (int i = 0; i < 4; ++i) t.f[i] = *(const float4*)(src + 4 * i);
}
__device__ __forceinline__ void zero_tile(Tile16& t) {
  #pragma unroll
  for (int i = 0; i < 4; ++i) { t.f[i].x = 0.f; t.f[i].y = 0.f; t.f[i].z = 0.f; t.f[i].w = 0.f; }
}
__device__ __forceinline__ void st_lds_bf16(const Tile16& t, unsigned* dst) {
  #pragma unroll
  for (int i = 0; i < 4; ++i) {
    dst[2 * i]     = pack2bf(t.f[i].x, t.f[i].y);
    dst[2 * i + 1] = pack2bf(t.f[i].z, t.f[i].w);
  }
}

// ---------------------------------------------------------------------------
// GEMM: out[M=2048, N] = A[M, K] * W[N, K]^T (+ bias) (+ GELU) (+ residual)
// Block: 128x128 C tile, 256 threads = 8 waves, each wave 32x64 (2x4 WMMA
// tiles). Double-buffered LDS staging, one barrier per 32-wide K-step.
// ---------------------------------------------------------------------------
__global__ __launch_bounds__(256, 1)
void gemm_kernel(const float* __restrict__ A, const float* __restrict__ W,
                 const float* __restrict__ bias, const float* __restrict__ resid,
                 float* __restrict__ out, int N, int K, int gelu)
{
  __shared__ unsigned short As[2][128 * LDSW];
  __shared__ unsigned short Ws[2][128 * LDSW];
  const int tid    = threadIdx.x;
  const int lane   = tid & 31;
  const int laneLo = lane & 15;
  const int laneHi = lane >> 4;
  const int wave   = tid >> 5;
  const int m0 = blockIdx.y * 128;
  const int n0 = blockIdx.x * 128;
  const int wm = (wave >> 1) * 32;   // wave's M offset inside block tile
  const int wn = (wave & 1) * 64;    // wave's N offset inside block tile
  const int r  = tid >> 1;           // staging: row
  const int c0 = (tid & 1) * 16;     // staging: col chunk

  v8f acc[2][4];
  #pragma unroll
  for (int i = 0; i < 2; ++i)
    #pragma unroll
    for (int j = 0; j < 4; ++j)
      #pragma unroll
      for (int v = 0; v < 8; ++v) acc[i][j][v] = 0.f;

  const float* aptr  = A + (size_t)(m0 + r) * K + c0;
  const int    wr    = n0 + r;
  const bool   wvalid = wr < N;
  const float* wptr  = W + (size_t)wr * K + c0;

  // Prologue: stage tile 0 into buffer 0.
  {
    Tile16 ta, tw;
    ldg_tile(ta, aptr);
    if (wvalid) ldg_tile(tw, wptr); else zero_tile(tw);
    st_lds_bf16(ta, (unsigned*)&As[0][r * LDSW + c0]);
    st_lds_bf16(tw, (unsigned*)&Ws[0][r * LDSW + c0]);
  }
  __syncthreads();

  int buf = 0;
  for (int kt = 0; kt < K; kt += 32) {
    const bool more = (kt + 32) < K;
    Tile16 na, nw;
    if (more) {                       // prefetch tile k+1 (overlaps WMMAs)
      ldg_tile(na, aptr + kt + 32);
      if (wvalid) ldg_tile(nw, wptr + kt + 32); else zero_tile(nw);
    }

    // A fragments: lane=row (M), element e -> k = e + 16*laneHi
    Frag af[2];
    #pragma unroll
    for (int i = 0; i < 2; ++i) {
      const unsigned short* base = &As[buf][(wm + i * 16 + laneLo) * LDSW];
      const int kb = laneHi * 16;
      af[i].q[0] = *(const v4u*)(base + kb);
      af[i].q[1] = *(const v4u*)(base + kb + 4);
      af[i].q[2] = *(const v4u*)(base + kb + 8);
      af[i].q[3] = *(const v4u*)(base + kb + 12);
    }
    // B fragments: lane=col (N = W row), element e -> k = e + 16*laneHi
    Frag bfr[4];
    #pragma unroll
    for (int j = 0; j < 4; ++j) {
      const unsigned short* base = &Ws[buf][(wn + j * 16 + laneLo) * LDSW + laneHi * 16];
      bfr[j].q[0] = *(const v4u*)(base);
      bfr[j].q[1] = *(const v4u*)(base + 4);
      bfr[j].q[2] = *(const v4u*)(base + 8);
      bfr[j].q[3] = *(const v4u*)(base + 12);
    }
    #pragma unroll
    for (int i = 0; i < 2; ++i)
      #pragma unroll
      for (int j = 0; j < 4; ++j)
        acc[i][j] = wmma_bf16(af[i], bfr[j], acc[i][j]);

    if (more) {                       // fill the other buffer, then one barrier
      st_lds_bf16(na, (unsigned*)&As[buf ^ 1][r * LDSW + c0]);
      st_lds_bf16(nw, (unsigned*)&Ws[buf ^ 1][r * LDSW + c0]);
    }
    __syncthreads();
    buf ^= 1;
  }

  // Epilogue: C layout -> lane=col n, VGPR v -> row m = v + 8*laneHi
  #pragma unroll
  for (int i = 0; i < 2; ++i) {
    #pragma unroll
    for (int j = 0; j < 4; ++j) {
      const int col = n0 + wn + j * 16 + laneLo;
      if (col >= N) continue;
      const float bv = bias ? bias[col] : 0.f;
      #pragma unroll
      for (int v = 0; v < 8; ++v) {
        const int row = m0 + wm + i * 16 + v + 8 * laneHi;
        float xv = acc[i][j][v] + bv;
        if (gelu) xv = 0.5f * xv * (1.f + erff(xv * 0.70710678118f));
        const size_t o = (size_t)row * N + col;
        if (resid) xv += resid[o];
        out[o] = xv;
      }
    }
  }
}

// ---------------------------------------------------------------------------
// Flash-style causal attention. qkv: [B*T, 3*DM] (q|k|v). y: [B*T, DM].
// Block = 128 q rows, 8 waves x 16-row q-tiles. Per 32-key block:
// 4 WMMAs for S = Q*K^T, online softmax, 4 WMMAs for O += P*V.
// ---------------------------------------------------------------------------
__global__ __launch_bounds__(256, 1)
void attn_kernel(const float* __restrict__ qkv, float* __restrict__ y)
{
  __shared__ unsigned short Pl[8 * 16 * LDSW];  // per-wave 16x32 bf16 P tile
  const int tid    = threadIdx.x;
  const int lane   = tid & 31;
  const int laneLo = lane & 15;
  const int laneHi = lane >> 4;
  const int wave   = tid >> 5;
  const int b  = blockIdx.z;
  const int h  = blockIdx.y;
  const int q0 = blockIdx.x * 128 + wave * 16;
  const float* base = qkv + (size_t)b * TS * (3 * DM) + h * HDIM;
  unsigned short* Pw = &Pl[wave * 16 * LDSW];

  // Q fragments for K-steps s=0,1 (d = s*32 + frag pattern); reused all j.
  Frag qf[2];
  {
    const float* qr = base + (size_t)(q0 + laneLo) * (3 * DM);
    #pragma unroll
    for (int s = 0; s < 2; ++s) {
      const int d0 = s * 32 + laneHi * 16;
      float4 f0 = *(const float4*)(qr + d0);
      float4 f1 = *(const float4*)(qr + d0 + 4);
      float4 f2 = *(const float4*)(qr + d0 + 8);
      float4 f3 = *(const float4*)(qr + d0 + 12);
      qf[s].s[0]  = f2bf(f0.x); qf[s].s[1]  = f2bf(f0.y);
      qf[s].s[2]  = f2bf(f0.z); qf[s].s[3]  = f2bf(f0.w);
      qf[s].s[4]  = f2bf(f1.x); qf[s].s[5]  = f2bf(f1.y);
      qf[s].s[6]  = f2bf(f1.z); qf[s].s[7]  = f2bf(f1.w);
      qf[s].s[8]  = f2bf(f2.x); qf[s].s[9]  = f2bf(f2.y);
      qf[s].s[10] = f2bf(f2.z); qf[s].s[11] = f2bf(f2.w);
      qf[s].s[12] = f2bf(f3.x); qf[s].s[13] = f2bf(f3.y);
      qf[s].s[14] = f2bf(f3.z); qf[s].s[15] = f2bf(f3.w);
    }
  }

  v8f o[4];
  #pragma unroll
  for (int j = 0; j < 4; ++j)
    #pragma unroll
    for (int v = 0; v < 8; ++v) o[j][v] = 0.f;
  float mrun[8], lrun[8];
  #pragma unroll
  for (int v = 0; v < 8; ++v) { mrun[v] = -3.0e38f; lrun[v] = 0.f; }

  const int qmax = q0 + 15;
  for (int jb = 0; jb <= qmax; jb += 32) {
    v8f S[2];
    #pragma unroll
    for (int u = 0; u < 2; ++u)
      #pragma unroll
      for (int v = 0; v < 8; ++v) S[u][v] = 0.f;

    // S = Q * K^T : B fragment = contiguous 16 floats of K row (lane = key)
    #pragma unroll
    for (int u = 0; u < 2; ++u) {
      const int key = jb + u * 16 + laneLo;
      const int kc  = key < TS ? key : TS - 1;
      const float* kr = base + (size_t)kc * (3 * DM) + DM;
      #pragma unroll
      for (int s = 0; s < 2; ++s) {
        Frag kf;
        const int d0 = s * 32 + laneHi * 16;
        float4 f0 = *(const float4*)(kr + d0);
        float4 f1 = *(const float4*)(kr + d0 + 4);
        float4 f2 = *(const float4*)(kr + d0 + 8);
        float4 f3 = *(const float4*)(kr + d0 + 12);
        kf.s[0]  = f2bf(f0.x); kf.s[1]  = f2bf(f0.y);
        kf.s[2]  = f2bf(f0.z); kf.s[3]  = f2bf(f0.w);
        kf.s[4]  = f2bf(f1.x); kf.s[5]  = f2bf(f1.y);
        kf.s[6]  = f2bf(f1.z); kf.s[7]  = f2bf(f1.w);
        kf.s[8]  = f2bf(f2.x); kf.s[9]  = f2bf(f2.y);
        kf.s[10] = f2bf(f2.z); kf.s[11] = f2bf(f2.w);
        kf.s[12] = f2bf(f3.x); kf.s[13] = f2bf(f3.y);
        kf.s[14] = f2bf(f3.z); kf.s[15] = f2bf(f3.w);
        S[u] = wmma_bf16(qf[s], kf, S[u]);
      }
    }

    // Online softmax (rows live at VGPR v, 16-lane halves; xor<=8 stays in half)
    #pragma unroll
    for (int v = 0; v < 8; ++v) {
      const int qr_abs = q0 + v + 8 * laneHi;
      float a0 = S[0][v] * 0.125f;
      float a1 = S[1][v] * 0.125f;
      a0 = (jb + laneLo > qr_abs)      ? -3.0e38f : a0;
      a1 = (jb + 16 + laneLo > qr_abs) ? -3.0e38f : a1;
      float t = fmaxf(a0, a1);
      t = fmaxf(t, __shfl_xor(t, 1));
      t = fmaxf(t, __shfl_xor(t, 2));
      t = fmaxf(t, __shfl_xor(t, 4));
      t = fmaxf(t, __shfl_xor(t, 8));
      const float mnew = fmaxf(mrun[v], t);
      const float scl  = __expf(mrun[v] - mnew);
      const float p0   = __expf(a0 - mnew);
      const float p1   = __expf(a1 - mnew);
      float rs = p0 + p1;
      rs += __shfl_xor(rs, 1);
      rs += __shfl_xor(rs, 2);
      rs += __shfl_xor(rs, 4);
      rs += __shfl_xor(rs, 8);
      lrun[v] = lrun[v] * scl + rs;
      mrun[v] = mnew;
      o[0][v] *= scl; o[1][v] *= scl; o[2][v] *= scl; o[3][v] *= scl;
      Pw[(v + 8 * laneHi) * LDSW + laneLo]      = f2bf(p0);
      Pw[(v + 8 * laneHi) * LDSW + 16 + laneLo] = f2bf(p1);
    }
    asm volatile("s_wait_dscnt 0" ::: "memory");   // C-layout -> A-layout via LDS

    Frag pf;
    {
      const unsigned short* pb = Pw + laneLo * LDSW;
      const int kb = laneHi * 16;
      pf.q[0] = *(const v4u*)(pb + kb);
      pf.q[1] = *(const v4u*)(pb + kb + 4);
      pf.q[2] = *(const v4u*)(pb + kb + 8);
      pf.q[3] = *(const v4u*)(pb + kb + 12);
    }
    // O += P * V : B fragment lane = d col, elements = 16 strided keys
    #pragma unroll
    for (int ds = 0; ds < 4; ++ds) {
      Frag vf;
      const int n = ds * 16 + laneLo;
      #pragma unroll
      for (int e = 0; e < 16; ++e) {
        int key = jb + e + 16 * laneHi;
        key = key < TS ? key : TS - 1;
        vf.s[e] = f2bf(base[(size_t)key * (3 * DM) + 2 * DM + n]);
      }
      o[ds] = wmma_bf16(pf, vf, o[ds]);
    }
  }

  #pragma unroll
  for (int ds = 0; ds < 4; ++ds)
    #pragma unroll
    for (int v = 0; v < 8; ++v) {
      const int row = q0 + v + 8 * laneHi;
      y[((size_t)b * TS + row) * DM + h * HDIM + ds * 16 + laneLo] =
          o[ds][v] / lrun[v];
    }
}

// ---------------------------------------------------------------------------
// LayerNorm, one block per row of 1024.
// ---------------------------------------------------------------------------
__global__ __launch_bounds__(256)
void ln_kernel(const float* __restrict__ x, const float* __restrict__ w,
               const float* __restrict__ bvec, float* __restrict__ out)
{
  __shared__ float r1[256], r2[256];
  const int tid = threadIdx.x;
  const float* xr = x + (size_t)blockIdx.x * DM;
  float lv[4], s = 0.f, sq = 0.f;
  #pragma unroll
  for (int i = 0; i < 4; ++i) {
    lv[i] = xr[tid + 256 * i];
    s  += lv[i];
    sq += lv[i] * lv[i];
  }
  r1[tid] = s; r2[tid] = sq;
  __syncthreads();
  for (int off = 128; off > 0; off >>= 1) {
    if (tid < off) { r1[tid] += r1[tid + off]; r2[tid] += r2[tid + off]; }
    __syncthreads();
  }
  const float mean = r1[0] * (1.f / DM);
  const float var  = r2[0] * (1.f / DM) - mean * mean;
  const float rstd = rsqrtf(var + 1e-5f);
  float* orow = out + (size_t)blockIdx.x * DM;
  #pragma unroll
  for (int i = 0; i < 4; ++i) {
    const int d = tid + 256 * i;
    orow[d] = (lv[i] - mean) * rstd * w[d] + bvec[d];
  }
}

// ---------------------------------------------------------------------------
// Embedding: x[row] = wte[idx[row]] + wpe[row % T]
// ---------------------------------------------------------------------------
__global__ __launch_bounds__(256)
void embed_kernel(const int* __restrict__ idx, const float* __restrict__ wte,
                  const float* __restrict__ wpe, float* __restrict__ x)
{
  const int row = blockIdx.x;
  const int t   = row & (TS - 1);
  const float* e = wte + (size_t)idx[row] * DM;
  const float* p = wpe + (size_t)t * DM;
  float* o = x + (size_t)row * DM;
  for (int d = threadIdx.x; d < DM; d += 256) o[d] = e[d] + p[d];
}

__global__ void zero_loss_kernel(float* loss) {
  if (threadIdx.x == 0 && blockIdx.x == 0) *loss = 0.f;
}

// ---------------------------------------------------------------------------
// Loss: per-row online logsumexp over V, -logp[target], mean via atomicAdd.
// ---------------------------------------------------------------------------
__global__ __launch_bounds__(256)
void loss_kernel(const float* __restrict__ logits, const int* __restrict__ targets,
                 float* __restrict__ loss)
{
  __shared__ float mred[256], sred[256];
  const int tid = threadIdx.x;
  const int row = blockIdx.x;
  const float* lr = logits + (size_t)row * VOC;
  float m = -3.0e38f, s = 0.f;
  for (int v = tid; v < VOC; v += 256) {
    const float x = lr[v];
    if (x > m) { s = s * __expf(m - x) + 1.f; m = x; }
    else       { s += __expf(x - m); }
  }
  mred[tid] = m; sred[tid] = s;
  __syncthreads();
  for (int off = 128; off > 0; off >>= 1) {
    if (tid < off) {
      const float m2 = mred[tid + off], s2 = sred[tid + off];
      const float mm = fmaxf(mred[tid], m2);
      sred[tid] = sred[tid] * __expf(mred[tid] - mm) + s2 * __expf(m2 - mm);
      mred[tid] = mm;
    }
    __syncthreads();
  }
  if (tid == 0) {
    const float lse = mred[0] + __logf(sred[0]);
    const float lt  = lr[targets[row]];
    atomicAdd(loss, (lse - lt) * (1.f / (float)MR));
  }
}

// ---------------------------------------------------------------------------
extern "C" void kernel_launch(void* const* d_in, const int* in_sizes, int n_in,
                              void* d_out, int out_size, void* d_ws, size_t ws_size,
                              hipStream_t stream) {
  (void)in_sizes; (void)n_in; (void)out_size; (void)ws_size;
  const int*   idx     = (const int*)d_in[0];
  const int*   targets = (const int*)d_in[1];
  const float* wte     = (const float*)d_in[2];
  const float* wpe     = (const float*)d_in[3];
  const float* ln1_w   = (const float*)d_in[4];
  const float* ln1_b   = (const float*)d_in[5];
  const float* attn_w  = (const float*)d_in[6];
  const float* attn_b  = (const float*)d_in[7];
  const float* proj_w  = (const float*)d_in[8];
  const float* proj_b  = (const float*)d_in[9];
  const float* ln2_w   = (const float*)d_in[10];
  const float* ln2_b   = (const float*)d_in[11];
  const float* fc_w    = (const float*)d_in[12];
  const float* fc_b    = (const float*)d_in[13];
  const float* fcp_w   = (const float*)d_in[14];
  const float* fcp_b   = (const float*)d_in[15];
  const float* lnf_w   = (const float*)d_in[16];
  const float* lnf_b   = (const float*)d_in[17];

  float* ws  = (float*)d_ws;
  float* x   = ws;                                  // [2048,1024]
  float* h   = x   + (size_t)MR * DM;               // [2048,1024]
  float* qkv = h   + (size_t)MR * DM;               // [2048,3072]
  float* y   = qkv + (size_t)MR * 3 * DM;           // [2048,1024]
  float* h2  = y   + (size_t)MR * DM;               // [2048,4096]

  float* logits = (float*)d_out;                    // [2048,50257]
  float* loss   = logits + (size_t)MR * VOC;        // [1]

  const dim3 blk(256);
  zero_loss_kernel<<<dim3(1), dim3(64), 0, stream>>>(loss);
  embed_kernel<<<dim3(MR), blk, 0, stream>>>(idx, wte, wpe, x);

  for (int l = 0; l < NLAYER; ++l) {
    ln_kernel<<<dim3(MR), blk, 0, stream>>>(x, ln1_w + (size_t)l * DM,
                                            ln1_b + (size_t)l * DM, h);
    gemm_kernel<<<dim3(3 * DM / 128, MR / 128), blk, 0, stream>>>(
        h, attn_w + (size_t)l * 3 * DM * DM, attn_b + (size_t)l * 3 * DM,
        nullptr, qkv, 3 * DM, DM, 0);
    attn_kernel<<<dim3(TS / 128, NHEAD, NB), blk, 0, stream>>>(qkv, y);
    gemm_kernel<<<dim3(DM / 128, MR / 128), blk, 0, stream>>>(
        y, proj_w + (size_t)l * DM * DM, proj_b + (size_t)l * DM,
        x, x, DM, DM, 0);
    ln_kernel<<<dim3(MR), blk, 0, stream>>>(x, ln2_w + (size_t)l * DM,
                                            ln2_b + (size_t)l * DM, h);
    gemm_kernel<<<dim3(4 * DM / 128, MR / 128), blk, 0, stream>>>(
        h, fc_w + (size_t)l * 4 * DM * DM, fc_b + (size_t)l * 4 * DM,
        nullptr, h2, 4 * DM, DM, 1);
    gemm_kernel<<<dim3(DM / 128, MR / 128), blk, 0, stream>>>(
        h2, fcp_w + (size_t)l * DM * 4 * DM, fcp_b + (size_t)l * DM,
        x, x, DM, 4 * DM, 0);
  }

  ln_kernel<<<dim3(MR), blk, 0, stream>>>(x, lnf_w, lnf_b, h);
  gemm_kernel<<<dim3((VOC + 127) / 128, MR / 128), blk, 0, stream>>>(
      h, wte, nullptr, nullptr, logits, VOC, DM, 0);
  loss_kernel<<<dim3(MR), blk, 0, stream>>>(logits, targets, loss);
}
